// PointnetFPModule_55327768708594
// MI455X (gfx1250) — matrix operand order repaired and checked by reading, not compiled, not run on hardware
//
#include <hip/hip_runtime.h>
#include <hip/hip_bf16.h>
#include <stdint.h>

// ---------------------------------------------------------------------------
// PointNet++ Feature Propagation (three_nn + three_interpolate + 2x SharedMLP)
// MI455X / gfx1250: wave32, WMMA 16x16x32 bf16 (f32 accumulate) for the MLPs,
// Tensor Data Mover (tensor_load_to_lds) for LDS staging.
// ---------------------------------------------------------------------------

namespace {

constexpr int B  = 8;
constexpr int N  = 8192;   // unknown points
constexpr int M  = 2048;   // known points
constexpr int C1 = 128;    // unknown_feats channels
constexpr int C2 = 256;    // known_feats channels
constexpr int CH = C1 + C2;  // 384 = MLP1 input channels
constexpr int CO = 256;      // MLP1/MLP2 output channels

constexpr int PTS = 32;      // points per workgroup in the fused MLP kernel

// LDS tile pitches (in ushorts). pitch/2 ≡ 4 (mod 64) -> ds_read_b128 across
// rows hits distinct 4-bank groups (minimal LDS bank conflicts).
constexpr int XPITCH = 392;  // >= 384
constexpr int YPITCH = 264;  // >= 256

typedef __attribute__((ext_vector_type(16))) __bf16 v16bf;
typedef __attribute__((ext_vector_type(8)))  float  v8f;

typedef unsigned int u32x4 __attribute__((ext_vector_type(4)));
typedef int          i32x8 __attribute__((ext_vector_type(8)));
typedef int          i32x4 __attribute__((ext_vector_type(4)));

__device__ __forceinline__ unsigned short f2bf(float f) {
  // round-to-nearest-even f32 -> bf16
  unsigned u = __builtin_bit_cast(unsigned, f);
  u += 0x7FFFu + ((u >> 16) & 1u);
  return (unsigned short)(u >> 16);
}

union BFragU {
  v16bf v;
  uint4 q[2];
};

// ---------------------------------------------------------------------------
// Tensor Data Mover: D# descriptor per CDNA5 ISA.
//   group0: [1:0]=count=1, [63:32]=lds_addr, [120:64]=global_addr, [127:126]=2
//   group1: [17:16]=data_size(2 -> 4B), dims/tile/stride fields as below
// Toolchain arity: ROCm7.2/clang-22 = 5 args, clang-23 = 6 args.
// ---------------------------------------------------------------------------
#if defined(__gfx1250__) && __has_builtin(__builtin_amdgcn_tensor_load_to_lds) && \
    __has_builtin(__builtin_amdgcn_s_wait_tensorcnt)
#define HAVE_TDM 1
#endif

#ifdef HAVE_TDM
// Load a (tile_d1 x tile_d0) f32 tile whose rows are stride0 elements apart in
// global memory into contiguous LDS at lds_addr (row-major, no padding).
__device__ __forceinline__ void tdm_load_f32_2d(unsigned lds_addr, const void* gptr,
                                                unsigned tile_d0, unsigned tile_d1,
                                                unsigned tensor_d0, unsigned tensor_d1,
                                                unsigned stride0) {
  const unsigned long long ga = (unsigned long long)(uintptr_t)gptr;
  u32x4 g0;
  g0[0] = 1u;                                                  // count=1, flags=0
  g0[1] = lds_addr;                                            // LDS byte address
  g0[2] = (unsigned)ga;                                        // global_addr[31:0]
  g0[3] = (unsigned)((ga >> 32) & 0x1FFFFFFull) | (2u << 30);  // ga[56:32] | type=2
  i32x8 g1;
  g1[0] = (int)(2u << 16);                                     // data_size = 4 bytes
  g1[1] = (int)((tensor_d0 & 0xFFFFu) << 16);                  // abar=0 | dim0[15:0]
  g1[2] = (int)(((tensor_d0 >> 16) & 0xFFFFu) |
                ((tensor_d1 & 0xFFFFu) << 16));                // dim0[31:16] | dim1[15:0]
  g1[3] = (int)(((tensor_d1 >> 16) & 0xFFFFu) |
                ((tile_d0 & 0xFFFFu) << 16));                  // dim1[31:16] | tile_dim0
  g1[4] = (int)(tile_d1 & 0xFFFFu);                            // tile_dim1 | tile_dim2=0
  g1[5] = (int)stride0;                                        // dim0_stride[31:0]
  g1[6] = 0;                                                   // dim0_stride[47:32] | dim1_stride lo
  g1[7] = 0;
  const i32x4 z4 = {0, 0, 0, 0};
#if __clang_major__ >= 23
  const i32x8 z8 = {0, 0, 0, 0, 0, 0, 0, 0};
  __builtin_amdgcn_tensor_load_to_lds(g0, g1, z4, z4, z8, 0);
#else
  __builtin_amdgcn_tensor_load_to_lds(g0, g1, z4, z4, 0);
#endif
}

__device__ __forceinline__ unsigned lds_byte_addr(const void* p) {
  // addrspacecast(as3 -> flat) places the LDS byte offset in the low 32 bits
  return (unsigned)(uintptr_t)p;
}
#endif  // HAVE_TDM

// A-matrix fragment (16x32 bf16, one 16-row tile of the weight matrix).
// Layout (matches B frag): lane<16: M=lane, K=k0..k0+15;
// lane>=16: M=lane-16, K=k0+16..k0+31.
__device__ __forceinline__ v16bf load_a_frag(const unsigned short* __restrict__ W,
                                             int ldk, int m, int kbase) {
  BFragU u;
  const uint4* src = (const uint4*)(W + (size_t)m * ldk + kbase);
  u.q[0] = src[0];   // 8 bf16: K = kbase .. kbase+7
  u.q[1] = src[1];   // 8 bf16: K = kbase+8 .. kbase+15
  return u.v;
}

// B-matrix fragment (32x16 bf16) from an LDS tile stored point-major T[pt][k].
// lane<16: N=lane, K=k0..k0+15; lane>=16: N=lane-16, K=k0+16..k0+31.
__device__ __forceinline__ v16bf load_b_frag(const unsigned short* T,
                                             int pitch, int lane, int k0) {
  const int nn = lane & 15;
  const int kk = k0 + ((lane < 16) ? 0 : 16);
  BFragU u;
  const uint4* src = (const uint4*)(T + nn * pitch + kk);
  u.q[0] = src[0];
  u.q[1] = src[1];
  return u.v;
}

// One MLP stage for one wave: 32 out-channels x 32 points as a 2x2 grid of
// 16x16 WMMA tiles (each A-fragment feeds two WMMAs), then bias + ReLU.
template <int KDIM>
__device__ __forceinline__ void mlp_stage(const unsigned short* __restrict__ Wbf,
                                          const unsigned short* Btile, int pitch,
                                          const float* __restrict__ bias,
                                          int lane, int wave, v8f (&acc)[4]) {
  const int ob0 = wave * 32;
  const int ob1 = ob0 + 16;
  const int mlow = lane & 15;
  acc[0] = {}; acc[1] = {}; acc[2] = {}; acc[3] = {};
#pragma unroll
  for (int k0 = 0; k0 < KDIM; k0 += 32) {
    const int kbase = k0 + ((lane < 16) ? 0 : 16);
    v16bf b0 = load_b_frag(Btile, pitch, lane, k0);               // points 0..15
    v16bf b1 = load_b_frag(Btile + 16 * pitch, pitch, lane, k0);  // points 16..31
    v16bf a0 = load_a_frag(Wbf, KDIM, ob0 + mlow, kbase);
    v16bf a1 = load_a_frag(Wbf, KDIM, ob1 + mlow, kbase);
    acc[0] = __builtin_amdgcn_wmma_f32_16x16x32_bf16(false, a0, false, b0,
                                                     (short)0, acc[0], false, false);
    acc[1] = __builtin_amdgcn_wmma_f32_16x16x32_bf16(false, a0, false, b1,
                                                     (short)0, acc[1], false, false);
    acc[2] = __builtin_amdgcn_wmma_f32_16x16x32_bf16(false, a1, false, b0,
                                                     (short)0, acc[2], false, false);
    acc[3] = __builtin_amdgcn_wmma_f32_16x16x32_bf16(false, a1, false, b1,
                                                     (short)0, acc[3], false, false);
  }
  // C/D layout: VGPR r -> M = obase + r (lane<16) or obase + 8 + r (lane>=16)
  const int mb0 = ob0 + ((lane < 16) ? 0 : 8);
  const int mb1 = ob1 + ((lane < 16) ? 0 : 8);
#pragma unroll
  for (int r = 0; r < 8; ++r) {
    acc[0][r] = fmaxf(acc[0][r] + bias[mb0 + r], 0.0f);
    acc[1][r] = fmaxf(acc[1][r] + bias[mb0 + r], 0.0f);
    acc[2][r] = fmaxf(acc[2][r] + bias[mb1 + r], 0.0f);
    acc[3][r] = fmaxf(acc[3][r] + bias[mb1 + r], 0.0f);
  }
}

}  // namespace

// ---------------------------------------------------------------------------
// Kernel 0: convert W1/W2 to bf16 once per launch (deterministic).
// ---------------------------------------------------------------------------
__global__ void convert_weights_kernel(const float* __restrict__ W1,
                                       const float* __restrict__ W2,
                                       unsigned short* __restrict__ o1,
                                       unsigned short* __restrict__ o2) {
  int i = blockIdx.x * blockDim.x + threadIdx.x;  // grid sized exactly
  if (i < CO * CH) {
    o1[i] = f2bf(W1[i]);
  } else {
    int j = i - CO * CH;
    o2[j] = f2bf(W2[j]);
  }
}

// ---------------------------------------------------------------------------
// Kernel 1: three_nn + inverse-distance weights.
// One thread per query point; all 2048 known xyz staged in LDS via the TDM
// (one DMA descriptor, 24 KB), inner-loop LDS reads are wave-uniform.
// ---------------------------------------------------------------------------
__global__ void three_nn_kernel(const float* __restrict__ unknown,  // [B,N,3]
                                const float* __restrict__ known,    // [B,M,3]
                                int* __restrict__ out_idx,          // [B,N,3]
                                float* __restrict__ out_w) {        // [B,N,3]
  __shared__ __align__(16) float s_k[M * 3];

  const int b = blockIdx.y;
  const float* kb = known + (size_t)b * M * 3;
#ifdef HAVE_TDM
  if (threadIdx.x == 0) {
    tdm_load_f32_2d(lds_byte_addr(s_k), kb, /*tile_d0=*/M * 3, /*tile_d1=*/1,
                    /*tensor_d0=*/M * 3, /*tensor_d1=*/1, /*stride0=*/M * 3);
  }
  __builtin_amdgcn_s_wait_tensorcnt(0);  // no-op for non-issuing waves
#else
  for (int t = threadIdx.x; t < M * 3; t += blockDim.x) s_k[t] = kb[t];
#endif
  __syncthreads();

  const int p = blockIdx.x * blockDim.x + threadIdx.x;
  const float* up = unknown + ((size_t)b * N + p) * 3;
  const float ux = up[0], uy = up[1], uz = up[2];

  float d0 = 3.0e38f, d1 = 3.0e38f, d2 = 3.0e38f;
  int i0 = 0, i1 = 0, i2 = 0;

  for (int i = 0; i < M; ++i) {
    const float dx = ux - s_k[3 * i + 0];
    const float dy = uy - s_k[3 * i + 1];
    const float dz = uz - s_k[3 * i + 2];
    const float d = dx * dx + dy * dy + dz * dz;
    // strict '<' keeps the earliest index on ties, matching lax.top_k
    if (d < d2) {
      if (d < d1) {
        if (d < d0) {
          d2 = d1; i2 = i1; d1 = d0; i1 = i0; d0 = d; i0 = i;
        } else {
          d2 = d1; i2 = i1; d1 = d; i1 = i;
        }
      } else {
        d2 = d; i2 = i;
      }
    }
  }

  const float r0 = 1.0f / (sqrtf(d0) + 1e-8f);
  const float r1 = 1.0f / (sqrtf(d1) + 1e-8f);
  const float r2 = 1.0f / (sqrtf(d2) + 1e-8f);
  const float rs = 1.0f / (r0 + r1 + r2);

  const size_t g = ((size_t)b * N + p) * 3;
  out_idx[g + 0] = i0; out_idx[g + 1] = i1; out_idx[g + 2] = i2;
  out_w[g + 0] = r0 * rs; out_w[g + 1] = r1 * rs; out_w[g + 2] = r2 * rs;
}

// ---------------------------------------------------------------------------
// Kernel 2: fused gather-interpolate + concat + MLP1(ReLU) + MLP2(ReLU).
// 256 threads (8 wave32) per 32-point tile. Each wave owns 32 output channels
// as a 2x2 grid of 16x16 WMMA tiles. unknown_feats arrive via a 2-D TDM tile
// (128 rows x 32 f32, row stride N). Intermediates live in LDS only.
// ---------------------------------------------------------------------------
__global__ void __launch_bounds__(256)
fp_mlp_kernel(const float* __restrict__ unknown_feats,          // [B,C1,N]
              const float* __restrict__ known_feats,            // [B,C2,M]
              const int* __restrict__ idx3,                     // [B,N,3]
              const float* __restrict__ w3,                     // [B,N,3]
              const unsigned short* __restrict__ W1bf,          // [CO,CH] bf16
              const float* __restrict__ b1,                     // [CO]
              const unsigned short* __restrict__ W2bf,          // [CO,CO] bf16
              const float* __restrict__ b2,                     // [CO]
              float* __restrict__ out) {                        // [B,CO,N]
  __shared__ __align__(16) unsigned short Xt[PTS * XPITCH];  // 384-ch input tile
  __shared__ __align__(16) unsigned short Yt[PTS * YPITCH];  // 256-ch hidden tile
  __shared__ __align__(16) float Ut[C1 * PTS];               // TDM staging (f32)
  __shared__ int   sI[PTS * 3];
  __shared__ float sW[PTS * 3];

  const int b    = blockIdx.y;
  const int p0   = blockIdx.x * PTS;
  const int tid  = threadIdx.x;
  const int lane = tid & 31;
  const int wave = tid >> 5;

  const float* uf = unknown_feats + (size_t)b * C1 * N;

#ifdef HAVE_TDM
  // DMA the unknown_feats tile: C1 rows of PTS contiguous f32, row stride N.
  if (tid == 0) {
    tdm_load_f32_2d(lds_byte_addr(Ut), uf + p0, /*tile_d0=*/PTS, /*tile_d1=*/C1,
                    /*tensor_d0=*/N, /*tensor_d1=*/C1, /*stride0=*/N);
  }
#endif

  if (tid < PTS * 3) {
    const size_t g = ((size_t)b * N + p0) * 3 + tid;
    sI[tid] = idx3[g];
    sW[tid] = w3[g];
  }
  __syncthreads();

  // --- build X tile: rows = PTS points, cols = 384 channels (bf16) ----------
  // interpolated known_feats -> channels [0, 256)      (L2-resident gathers)
  {
    const float* kf = known_feats + (size_t)b * C2 * M + (size_t)tid * M;  // c = tid
#pragma unroll 4
    for (int p = 0; p < PTS; ++p) {
      const float v = sW[p * 3 + 0] * kf[sI[p * 3 + 0]] +
                      sW[p * 3 + 1] * kf[sI[p * 3 + 1]] +
                      sW[p * 3 + 2] * kf[sI[p * 3 + 2]];
      Xt[p * XPITCH + tid] = f2bf(v);
    }
  }
  // unknown_feats -> channels [256, 384)
#ifdef HAVE_TDM
  __builtin_amdgcn_s_wait_tensorcnt(0);
  __syncthreads();  // Ut visible to all waves
  for (int e = tid; e < C1 * PTS; e += 256) {
    const int c = e >> 5;        // Ut is [c][p], read linearly (conflict-free)
    const int p = e & (PTS - 1);
    Xt[p * XPITCH + C2 + c] = f2bf(Ut[e]);
  }
#else
  for (int e = tid; e < C1 * PTS; e += 256) {
    const int c = e >> 5;
    const int p = e & (PTS - 1);
    Xt[p * XPITCH + C2 + c] = f2bf(uf[(size_t)c * N + p0 + p]);
  }
#endif
  __syncthreads();

  // --- MLP1: (256 x 384) bf16 WMMA, bias + ReLU -----------------------------
  v8f acc[4];
  mlp_stage<CH>(W1bf, Xt, XPITCH, b1, lane, wave, acc);

  // spill hidden activations to LDS as bf16 (point-major for B-frag reloads)
  {
    const int nn0 = lane & 15;
    const int nn1 = nn0 + 16;
    const int mb0 = wave * 32 + ((lane < 16) ? 0 : 8);
    const int mb1 = wave * 32 + 16 + ((lane < 16) ? 0 : 8);
#pragma unroll
    for (int r = 0; r < 8; ++r) {
      Yt[nn0 * YPITCH + mb0 + r] = f2bf(acc[0][r]);
      Yt[nn1 * YPITCH + mb0 + r] = f2bf(acc[1][r]);
      Yt[nn0 * YPITCH + mb1 + r] = f2bf(acc[2][r]);
      Yt[nn1 * YPITCH + mb1 + r] = f2bf(acc[3][r]);
    }
  }
  __syncthreads();

  // --- MLP2: (256 x 256) bf16 WMMA, bias + ReLU -----------------------------
  mlp_stage<CO>(W2bf, Yt, YPITCH, b2, lane, wave, acc);

  // --- store: out[b][m][p0 + n]; 16 consecutive lanes -> 64B per row --------
  {
    const int nn0 = lane & 15;
    const int nn1 = nn0 + 16;
    const int mb0 = wave * 32 + ((lane < 16) ? 0 : 8);
    const int mb1 = wave * 32 + 16 + ((lane < 16) ? 0 : 8);
    float* ob = out + (size_t)b * CO * N + p0;
#pragma unroll
    for (int r = 0; r < 8; ++r) {
      ob[(size_t)(mb0 + r) * N + nn0] = acc[0][r];
      ob[(size_t)(mb0 + r) * N + nn1] = acc[1][r];
      ob[(size_t)(mb1 + r) * N + nn0] = acc[2][r];
      ob[(size_t)(mb1 + r) * N + nn1] = acc[3][r];
    }
  }
}

// ---------------------------------------------------------------------------
// Host-side launcher. Inputs (setup_inputs order):
//   0 unknown [B,N,3] f32, 1 known [B,M,3] f32, 2 unknown_feats [B,C1,N] f32,
//   3 known_feats [B,C2,M] f32, 4 W1 [256,384] f32, 5 b1 [256] f32,
//   6 W2 [256,256] f32, 7 b2 [256] f32.  Output: [B,256,N] f32.
// ---------------------------------------------------------------------------
extern "C" void kernel_launch(void* const* d_in, const int* in_sizes, int n_in,
                              void* d_out, int out_size, void* d_ws, size_t ws_size,
                              hipStream_t stream) {
  const float* unknown = (const float*)d_in[0];
  const float* known   = (const float*)d_in[1];
  const float* uf      = (const float*)d_in[2];
  const float* kf      = (const float*)d_in[3];
  const float* W1      = (const float*)d_in[4];
  const float* b1      = (const float*)d_in[5];
  const float* W2      = (const float*)d_in[6];
  const float* b2      = (const float*)d_in[7];
  float* out = (float*)d_out;

  // workspace layout (~1.9 MB total)
  char* ws = (char*)d_ws;
  int*            ws_idx = (int*)ws;                                 // B*N*3 int
  float*          ws_w   = (float*)(ws + (size_t)B * N * 3 * 4);     // B*N*3 f32
  unsigned short* w1bf   = (unsigned short*)(ws + (size_t)B * N * 3 * 8);
  unsigned short* w2bf   = w1bf + (size_t)CO * CH;

  // 1) weights -> bf16 (163840 elems, exact grid)
  convert_weights_kernel<<<dim3((CO * CH + CO * CO) / 256), 256, 0, stream>>>(
      W1, W2, w1bf, w2bf);

  // 2) three_nn + weights: one thread per query point
  three_nn_kernel<<<dim3(N / 256, B), 256, 0, stream>>>(unknown, known,
                                                        ws_idx, ws_w);

  // 3) fused interpolate + concat + MLP1 + MLP2 (WMMA)
  fp_mlp_kernel<<<dim3(N / PTS, B), 256, 0, stream>>>(uf, kf, ws_idx, ws_w,
                                                      w1bf, b1, w2bf, b2, out);
}
